// Transformer_63316407878396
// MI455X (gfx1250) — matrix operand order repaired and checked by reading, not compiled, not run on hardware
//
#include <hip/hip_runtime.h>
#include <hip/hip_bf16.h>
#include <math.h>

// ---- problem dims (fixed by reference) ----
#define Bb 4
#define Ss 128
#define Dd 512
#define Hh 8
#define DK 64
#define Ff 2048
#define Ll 2
#define Vv 32000
#define Nn (Bb*Ss)   // 512 nodes per side

typedef __attribute__((ext_vector_type(16))) __bf16 v16bf;
typedef __attribute__((ext_vector_type(8)))  __bf16 v8bf;
typedef __attribute__((ext_vector_type(8)))  float  v8f;

// ------------------------------------------------------------------
// Convert+transpose: W f32 [K, Nc] -> Wt bf16 [Nc, K]
// ------------------------------------------------------------------
__global__ void __launch_bounds__(256) k_transpose_bf16(
    const float* __restrict__ W, __bf16* __restrict__ Wt, int K, int Nc)
{
    int idx = blockIdx.x * 256 + threadIdx.x;
    if (idx >= K * Nc) return;
    int k = idx / Nc;
    int n = idx - k * Nc;
    Wt[(size_t)n * K + k] = (__bf16)W[idx];
}

// ------------------------------------------------------------------
// Embedding + sinusoidal positional encoding:
// x[n,d] = emb[tok[n],d]*sqrt(D) + pe(pos=n%S, d)
// ------------------------------------------------------------------
__global__ void __launch_bounds__(128) k_embed(
    const float* __restrict__ emb, const int* __restrict__ tok,
    float* __restrict__ x)
{
    int n   = blockIdx.x;
    int tid = threadIdx.x;
    int t   = tok[n];
    int pos = n % Ss;
    const float scale   = 22.62741699796952f;  // sqrt(512)
    const float ln10000 = 9.210340371976184f;
#pragma unroll
    for (int i = 0; i < Dd / 128; ++i) {
        int d  = tid + i * 128;
        int tt = d >> 1;
        float div = __expf(-ln10000 * (2.0f * (float)tt) / (float)Dd);
        float ang = (float)pos * div;
        float pe  = (d & 1) ? cosf(ang) : sinf(ang);
        x[(size_t)n * Dd + d] = emb[(size_t)t * Dd + d] * scale + pe;
    }
}

// ------------------------------------------------------------------
// LayerNorm over rows of [Nn, Dd]; gb = [gamma(D), beta(D)]; out bf16
// ------------------------------------------------------------------
__global__ void __launch_bounds__(128) k_ln(
    const float* __restrict__ x, const float* __restrict__ gb,
    __bf16* __restrict__ out)
{
    __shared__ float red[128];
    int r = blockIdx.x, tid = threadIdx.x;
    const float* xr = x + (size_t)r * Dd;
    float s = 0.f, s2 = 0.f;
#pragma unroll
    for (int i = 0; i < Dd / 128; ++i) {
        float v = xr[tid + i * 128];
        s += v; s2 += v * v;
    }
    red[tid] = s; __syncthreads();
    for (int off = 64; off > 0; off >>= 1) {
        if (tid < off) red[tid] += red[tid + off];
        __syncthreads();
    }
    float mu = red[0] / (float)Dd;
    __syncthreads();
    red[tid] = s2; __syncthreads();
    for (int off = 64; off > 0; off >>= 1) {
        if (tid < off) red[tid] += red[tid + off];
        __syncthreads();
    }
    float var  = red[0] / (float)Dd - mu * mu;
    float rstd = rsqrtf(var + 1e-5f);
#pragma unroll
    for (int i = 0; i < Dd / 128; ++i) {
        int d = tid + i * 128;
        float v = (xr[d] - mu) * rstd * gb[d] + gb[Dd + d];
        out[(size_t)r * Dd + d] = (__bf16)v;
    }
}

// ------------------------------------------------------------------
// WMMA GEMM: C[M,Nc] = A[M,K](bf16) * Bt[Nc,K](bf16, pre-transposed)
// One wave computes a 32x64 tile: 2 A-frags x 4 B-frags = 8x
// v_wmma_f32_16x16x32_bf16 per 32-wide K-step; each B fragment is
// reused by both A fragments. WGP-scope global_prefetch for the next
// K panel (weights are L2-resident; pull into near caches).
// Epilogue: optional bias (f32[Nc]), ReLU, residual (f32[M,Nc]),
//           output either f32 (Cf) or bf16 (Cb).
// Requires: M % 32 == 0, Nc % 64 == 0, K % 32 == 0.
// ------------------------------------------------------------------
__global__ void __launch_bounds__(32) k_gemm_bf16(
    const __bf16* __restrict__ A, const __bf16* __restrict__ Bt,
    int M, int Nc, int K,
    const float* __restrict__ bias, const float* __restrict__ resid,
    float* __restrict__ Cf, __bf16* __restrict__ Cb, int relu)
{
    const int lane = threadIdx.x;          // 0..31
    const int mt   = blockIdx.y;           // 32-row tile
    const int nb   = blockIdx.x * 64;      // 64-col base
    const int l15  = lane & 15;
    const int half = lane >> 4;            // 0 or 1
    const int ncol = nb + l15;
    const __bf16* arow0 = A + (size_t)(mt * 32 + l15) * K;
    const __bf16* arow1 = A + (size_t)(mt * 32 + 16 + l15) * K;

    v8f acc[2][4] = {};
    for (int k0 = 0; k0 < K; k0 += 32) {
        // WGP-scope prefetch of next K panel (locality 3 -> near caches)
        if (k0 + 32 < K) {
            __builtin_prefetch(arow0 + k0 + 32, 0, 3);
            __builtin_prefetch(arow1 + k0 + 32, 0, 3);
            __builtin_prefetch(Bt + (size_t)ncol * K + k0 + 32 + half * 16, 0, 3);
            __builtin_prefetch(Bt + (size_t)(ncol + 48) * K + k0 + 32 + half * 16, 0, 3);
        }
        // A fragments (16x32 bf16 layout):
        // lane<16 -> K[0..15]; lane>=16 -> K[16..31]
        v16bf a0, a1;
        {
            v8bf lo = *(const v8bf*)(arow0 + k0 + half * 16);
            v8bf hi = *(const v8bf*)(arow0 + k0 + 8 + half * 16);
#pragma unroll
            for (int i = 0; i < 8; ++i) { a0[i] = lo[i]; a0[8 + i] = hi[i]; }
            lo = *(const v8bf*)(arow1 + k0 + half * 16);
            hi = *(const v8bf*)(arow1 + k0 + 8 + half * 16);
#pragma unroll
            for (int i = 0; i < 8; ++i) { a1[i] = lo[i]; a1[8 + i] = hi[i]; }
        }
        const int kb = k0 + half * 16;     // B frag: lanes 0-15 K 0..15, lanes 16-31 K 16..31
#pragma unroll
        for (int t = 0; t < 4; ++t) {
            v16bf b = *(const v16bf*)(Bt + (size_t)(ncol + t * 16) * K + kb);
            acc[0][t] = __builtin_amdgcn_wmma_f32_16x16x32_bf16(
                false, a0, false, b, (short)0, acc[0][t], false, false);
            acc[1][t] = __builtin_amdgcn_wmma_f32_16x16x32_bf16(
                false, a1, false, b, (short)0, acc[1][t], false, false);
        }
    }

    // C/D layout: element i -> row = mt*32 + mi*16 + half*8 + i,
    //             col = nb + t*16 + l15
#pragma unroll
    for (int mi = 0; mi < 2; ++mi) {
#pragma unroll
        for (int t = 0; t < 4; ++t) {
            int n = ncol + t * 16;
            float bv = bias ? bias[n] : 0.f;
#pragma unroll
            for (int i = 0; i < 8; ++i) {
                int row = mt * 32 + mi * 16 + half * 8 + i;
                float v = acc[mi][t][i] + bv;
                if (relu) v = v > 0.f ? v : 0.f;
                if (resid) v += resid[(size_t)row * Nc + n];
                if (Cf) Cf[(size_t)row * Nc + n] = v;
                else    Cb[(size_t)row * Nc + n] = (__bf16)v;
            }
        }
    }
}

// ------------------------------------------------------------------
// Edge attention (dense per-batch block, optional causal):
// s = exp(clip(q.k/8, -10, 10)); out = sum(s*v)/sum(s); out bf16.
// One block of 128 threads per (dst node, head).
// ------------------------------------------------------------------
__global__ void __launch_bounds__(128) k_attn(
    const float* __restrict__ q, const float* __restrict__ k,
    const float* __restrict__ v, __bf16* __restrict__ o, int causal)
{
    __shared__ float qs[DK];
    __shared__ float sv[Ss];
    int bh  = blockIdx.x;
    int h   = bh % Hh;
    int dst = bh / Hh;
    int b   = dst / Ss;
    int pos = dst - b * Ss;
    int kvb = b * Ss;
    int tid = threadIdx.x;

    if (tid < DK) qs[tid] = q[(size_t)dst * Dd + h * DK + tid];
    __syncthreads();

    int limit = causal ? pos : (Ss - 1);
    float s = 0.f;
    if (tid <= limit) {
        const float* krow = k + (size_t)(kvb + tid) * Dd + h * DK;
        float dot = 0.f;
#pragma unroll 8
        for (int d = 0; d < DK; ++d) dot += krow[d] * qs[d];
        dot *= 0.125f;                       // 1/sqrt(64)
        dot = fminf(fmaxf(dot, -10.f), 10.f);
        s = __expf(dot);
    }
    sv[tid] = s;
    __syncthreads();

    if (tid < DK) {
        float z = 0.f, acc = 0.f;
        for (int j = 0; j <= limit; ++j) {
            float sj = sv[j];
            z   += sj;
            acc += sj * v[(size_t)(kvb + j) * Dd + h * DK + tid];
        }
        o[(size_t)dst * Dd + h * DK + tid] = (__bf16)(acc / z);
    }
}

// ------------------------------------------------------------------
// In-place row-wise log-softmax over [Nn, Vv]
// ------------------------------------------------------------------
__global__ void __launch_bounds__(256) k_logsoftmax(float* __restrict__ x)
{
    __shared__ float red[256];
    int r = blockIdx.x, tid = threadIdx.x;
    float* xr = x + (size_t)r * Vv;
    float m = -1e30f;
    for (int i = tid; i < Vv; i += 256) m = fmaxf(m, xr[i]);
    red[tid] = m; __syncthreads();
    for (int off = 128; off > 0; off >>= 1) {
        if (tid < off) red[tid] = fmaxf(red[tid], red[tid + off]);
        __syncthreads();
    }
    m = red[0]; __syncthreads();
    float s = 0.f;
    for (int i = tid; i < Vv; i += 256) s += __expf(xr[i] - m);
    red[tid] = s; __syncthreads();
    for (int off = 128; off > 0; off >>= 1) {
        if (tid < off) red[tid] += red[tid + off];
        __syncthreads();
    }
    float lse = m + logf(red[0]);
    for (int i = tid; i < Vv; i += 256) xr[i] -= lse;
}

// ==================================================================
extern "C" void kernel_launch(void* const* d_in, const int* in_sizes, int n_in,
                              void* d_out, int out_size, void* d_ws, size_t ws_size,
                              hipStream_t stream)
{
    (void)in_sizes; (void)n_in; (void)out_size; (void)ws_size;
    // ---- inputs (setup_inputs order) ----
    const float* src_emb  = (const float*)d_in[0];
    const float* tgt_emb  = (const float*)d_in[1];
    const float* enc_Wqkv = (const float*)d_in[2];   // [L,3,D,D]
    const float* enc_Wo   = (const float*)d_in[3];   // [L,D,D]
    const float* enc_ln1  = (const float*)d_in[4];   // [L,2,D]
    const float* enc_W1   = (const float*)d_in[5];   // [L,D,F]
    const float* enc_b1   = (const float*)d_in[6];   // [L,F]
    const float* enc_W2   = (const float*)d_in[7];   // [L,F,D]
    const float* enc_b2   = (const float*)d_in[8];   // [L,D]
    const float* enc_ln2  = (const float*)d_in[9];   // [L,2,D]
    const float* enc_lnf  = (const float*)d_in[10];  // [2,D]
    const float* dec_Wqkv = (const float*)d_in[11];
    const float* dec_Wo   = (const float*)d_in[12];
    const float* dec_ln1  = (const float*)d_in[13];
    const float* dec_cWq  = (const float*)d_in[14];
    const float* dec_cWkv = (const float*)d_in[15];  // [L,2,D,D]
    const float* dec_cWo  = (const float*)d_in[16];
    const float* dec_ln2  = (const float*)d_in[17];
    const float* dec_W1   = (const float*)d_in[18];
    const float* dec_b1   = (const float*)d_in[19];
    const float* dec_W2   = (const float*)d_in[20];
    const float* dec_b2   = (const float*)d_in[21];
    const float* dec_ln3  = (const float*)d_in[22];
    const float* dec_lnf  = (const float*)d_in[23];
    const float* gen_W    = (const float*)d_in[24];  // [D,V]
    const float* gen_b    = (const float*)d_in[25];  // [V]
    const int*   src_tok  = (const int*)d_in[26];
    const int*   tgt_tok  = (const int*)d_in[27];
    // edge lists (d_in[28..33]) are structurally known -> unused

    // ---- workspace carve-up ----
    char* wp = (char*)d_ws;
    auto alloc = [&](size_t bytes) -> void* {
        void* r = (void*)wp;
        wp += (bytes + 255) & ~(size_t)255;
        return r;
    };
    float*  x_e   = (float*)alloc((size_t)Nn * Dd * 4);
    float*  x_d   = (float*)alloc((size_t)Nn * Dd * 4);
    float*  qb    = (float*)alloc((size_t)Nn * Dd * 4);
    float*  kb    = (float*)alloc((size_t)Nn * Dd * 4);
    float*  vb    = (float*)alloc((size_t)Nn * Dd * 4);
    __bf16* xn    = (__bf16*)alloc((size_t)Nn * Dd * 2);
    __bf16* xe_b  = (__bf16*)alloc((size_t)Nn * Dd * 2);
    __bf16* o_b   = (__bf16*)alloc((size_t)Nn * Dd * 2);
    __bf16* hid_b = (__bf16*)alloc((size_t)Nn * Ff * 2);

    auto tposed = [&](const float* W, int K, int Nc) -> __bf16* {
        __bf16* Wt = (__bf16*)alloc((size_t)K * Nc * 2);
        int total = K * Nc;
        k_transpose_bf16<<<(total + 255) / 256, 256, 0, stream>>>(W, Wt, K, Nc);
        return Wt;
    };
    auto gemm = [&](const __bf16* A, const __bf16* Bt, int M, int Nc, int K,
                    const float* bias, const float* resid,
                    float* Cf, __bf16* Cb, int relu) {
        dim3 g(Nc / 64, M / 32);
        k_gemm_bf16<<<g, 32, 0, stream>>>(A, Bt, M, Nc, K, bias, resid, Cf, Cb, relu);
    };

    // ---- weight prep (f32 [K,N] -> bf16 [N,K]) ----
    __bf16 *eWq[Ll], *eWk[Ll], *eWv[Ll], *eWo[Ll], *eW1[Ll], *eW2[Ll];
    __bf16 *dWq[Ll], *dWk[Ll], *dWv[Ll], *dWo[Ll];
    __bf16 *cWq[Ll], *cWk[Ll], *cWv[Ll], *cWo[Ll], *dW1[Ll], *dW2[Ll];
    const size_t DD2 = (size_t)Dd * Dd;
    for (int l = 0; l < Ll; ++l) {
        eWq[l] = tposed(enc_Wqkv + (size_t)(l * 3 + 0) * DD2, Dd, Dd);
        eWk[l] = tposed(enc_Wqkv + (size_t)(l * 3 + 1) * DD2, Dd, Dd);
        eWv[l] = tposed(enc_Wqkv + (size_t)(l * 3 + 2) * DD2, Dd, Dd);
        eWo[l] = tposed(enc_Wo   + (size_t)l * DD2,           Dd, Dd);
        eW1[l] = tposed(enc_W1   + (size_t)l * Dd * Ff,       Dd, Ff);
        eW2[l] = tposed(enc_W2   + (size_t)l * Ff * Dd,       Ff, Dd);
        dWq[l] = tposed(dec_Wqkv + (size_t)(l * 3 + 0) * DD2, Dd, Dd);
        dWk[l] = tposed(dec_Wqkv + (size_t)(l * 3 + 1) * DD2, Dd, Dd);
        dWv[l] = tposed(dec_Wqkv + (size_t)(l * 3 + 2) * DD2, Dd, Dd);
        dWo[l] = tposed(dec_Wo   + (size_t)l * DD2,           Dd, Dd);
        cWq[l] = tposed(dec_cWq  + (size_t)l * DD2,           Dd, Dd);
        cWk[l] = tposed(dec_cWkv + (size_t)(l * 2 + 0) * DD2, Dd, Dd);
        cWv[l] = tposed(dec_cWkv + (size_t)(l * 2 + 1) * DD2, Dd, Dd);
        cWo[l] = tposed(dec_cWo  + (size_t)l * DD2,           Dd, Dd);
        dW1[l] = tposed(dec_W1   + (size_t)l * Dd * Ff,       Dd, Ff);
        dW2[l] = tposed(dec_W2   + (size_t)l * Ff * Dd,       Ff, Dd);
    }
    __bf16* genWt = tposed(gen_W, Dd, Vv);

    // ---- embeddings + positional encoding ----
    k_embed<<<Nn, 128, 0, stream>>>(src_emb, src_tok, x_e);
    k_embed<<<Nn, 128, 0, stream>>>(tgt_emb, tgt_tok, x_d);

    // ---- encoder ----
    for (int l = 0; l < Ll; ++l) {
        k_ln<<<Nn, 128, 0, stream>>>(x_e, enc_ln1 + (size_t)l * 2 * Dd, xn);
        gemm(xn, eWq[l], Nn, Dd, Dd, nullptr, nullptr, qb, nullptr, 0);
        gemm(xn, eWk[l], Nn, Dd, Dd, nullptr, nullptr, kb, nullptr, 0);
        gemm(xn, eWv[l], Nn, Dd, Dd, nullptr, nullptr, vb, nullptr, 0);
        k_attn<<<Nn * Hh, 128, 0, stream>>>(qb, kb, vb, o_b, 0);
        gemm(o_b, eWo[l], Nn, Dd, Dd, nullptr, x_e, x_e, nullptr, 0);
        k_ln<<<Nn, 128, 0, stream>>>(x_e, enc_ln2 + (size_t)l * 2 * Dd, xn);
        gemm(xn, eW1[l], Nn, Ff, Dd, enc_b1 + (size_t)l * Ff, nullptr, nullptr, hid_b, 1);
        gemm(hid_b, eW2[l], Nn, Dd, Ff, enc_b2 + (size_t)l * Dd, x_e, x_e, nullptr, 0);
    }
    k_ln<<<Nn, 128, 0, stream>>>(x_e, enc_lnf, xe_b);

    // ---- decoder ----
    for (int l = 0; l < Ll; ++l) {
        // causal self-attention
        k_ln<<<Nn, 128, 0, stream>>>(x_d, dec_ln1 + (size_t)l * 2 * Dd, xn);
        gemm(xn, dWq[l], Nn, Dd, Dd, nullptr, nullptr, qb, nullptr, 0);
        gemm(xn, dWk[l], Nn, Dd, Dd, nullptr, nullptr, kb, nullptr, 0);
        gemm(xn, dWv[l], Nn, Dd, Dd, nullptr, nullptr, vb, nullptr, 0);
        k_attn<<<Nn * Hh, 128, 0, stream>>>(qb, kb, vb, o_b, 1);
        gemm(o_b, dWo[l], Nn, Dd, Dd, nullptr, x_d, x_d, nullptr, 0);
        // cross-attention (K/V from final-LN'ed encoder output)
        k_ln<<<Nn, 128, 0, stream>>>(x_d, dec_ln2 + (size_t)l * 2 * Dd, xn);
        gemm(xn,   cWq[l], Nn, Dd, Dd, nullptr, nullptr, qb, nullptr, 0);
        gemm(xe_b, cWk[l], Nn, Dd, Dd, nullptr, nullptr, kb, nullptr, 0);
        gemm(xe_b, cWv[l], Nn, Dd, Dd, nullptr, nullptr, vb, nullptr, 0);
        k_attn<<<Nn * Hh, 128, 0, stream>>>(qb, kb, vb, o_b, 0);
        gemm(o_b, cWo[l], Nn, Dd, Dd, nullptr, x_d, x_d, nullptr, 0);
        // FFN
        k_ln<<<Nn, 128, 0, stream>>>(x_d, dec_ln3 + (size_t)l * 2 * Dd, xn);
        gemm(xn, dW1[l], Nn, Ff, Dd, dec_b1 + (size_t)l * Ff, nullptr, nullptr, hid_b, 1);
        gemm(hid_b, dW2[l], Nn, Dd, Ff, dec_b2 + (size_t)l * Dd, x_d, x_d, nullptr, 0);
    }

    // ---- generator + log-softmax ----
    k_ln<<<Nn, 128, 0, stream>>>(x_d, dec_lnf, xn);
    gemm(xn, genWt, Nn, Vv, Dd, gen_b, nullptr, (float*)d_out, nullptr, 0);
    k_logsoftmax<<<Nn, 256, 0, stream>>>((float*)d_out);
}
